// Codebook_1090921693417
// MI455X (gfx1250) — matrix operand order repaired and checked by reading, not compiled, not run on hardware
//
#include <hip/hip_runtime.h>
#include <hip/hip_bf16.h>

// Problem constants (from reference setup_inputs)
#define NN 32768
#define DD 512
#define KK 8192

typedef __bf16 v16bf __attribute__((ext_vector_type(16)));
typedef float  v8f   __attribute__((ext_vector_type(8)));

__device__ __forceinline__ __bf16 f32_to_bf16(float f) { return (__bf16)f; }

// ---------------------------------------------------------------------------
// Kernel 1: normalize codebook, emit bf16 hi/lo planes + bias = -0.5*||c||^2
// One wave (32 lanes) per code; 8 codes per 256-thread block.
// ---------------------------------------------------------------------------
__global__ void vq_prep_kernel(const float* __restrict__ codebook,
                               const float* __restrict__ counts,
                               __bf16* __restrict__ cb_hi,
                               __bf16* __restrict__ cb_lo,
                               float* __restrict__ bias) {
  const int k    = blockIdx.x * 8 + (threadIdx.x >> 5);
  const int lane = threadIdx.x & 31;
  const float inv = 1.0f / counts[k];
  const float* row = codebook + (size_t)k * DD;
  float ss = 0.0f;
  for (int d = lane; d < DD; d += 32) {
    const float c = row[d] * inv;
    ss += c * c;
    const __bf16 h = f32_to_bf16(c);
    cb_hi[(size_t)k * DD + d] = h;
    cb_lo[(size_t)k * DD + d] = f32_to_bf16(c - (float)h);
  }
#pragma unroll
  for (int off = 16; off > 0; off >>= 1) ss += __shfl_xor(ss, off, 32);
  if (lane == 0) bias[k] = -0.5f * ss;
}

// ---------------------------------------------------------------------------
// Kernel 2: fused GEMM (split-bf16, f32-accurate) + per-row argmax -> labels
//   score[n][k] = t[n].cb[k] - 0.5*||cb[k]||^2 ; label[n] = argmax_k score
// WG = 256 threads (8 waves), 64-row tile. Column register blocking cs=2:
// each wave owns two 16-col strips per N-tile (NT = 8 waves * 32 = 256 cols),
// so each A fragment (LDS) feeds 6 WMMAs -> LDS bytes/WMMA halved vs cs=1.
// A tile pre-swizzled to exact WMMA A-fragment order in 128KB dynamic LDS.
// ---------------------------------------------------------------------------
__global__ void __launch_bounds__(256)
vq_gemm_argmax_kernel(const float* __restrict__ target,
                      const __bf16* __restrict__ cb_hi,
                      const __bf16* __restrict__ cb_lo,
                      const float* __restrict__ bias,
                      int* __restrict__ labels) {
  constexpr int MT = 64;   // rows per WG
  constexpr int CS = 2;    // column strips per wave
  constexpr int NT = 32 * 8;  // codes per N-tile (8 waves * CS*16 cols)

  extern __shared__ char smem_raw[];
  __bf16* Ahi = (__bf16*)smem_raw;            // [MT*DD] fragment-ordered
  __bf16* Alo = (__bf16*)smem_raw + MT * DD;  // [MT*DD]
  __shared__ float redv[MT * 8];
  __shared__ int   redi[MT * 8];

  const int tid  = threadIdx.x;
  const int wave = tid >> 5;
  const int lane = tid & 31;
  const int half = lane >> 4;  // 0: lanes 0-15, 1: lanes 16-31
  const int mr   = lane & 15;
  const int m0   = blockIdx.x * MT;

  // ---- Stage A tile: f32 -> (bf16 hi, bf16 lo), stored in WMMA A-fragment
  // order: flat j = ((msub*16 + ks)*32 + lane)*16 + i
  //   row = msub*16 + (lane&15)
  //   k   = ks*32 + (lane>>4)*16 + i      (contiguous k, matches B)
  for (int j = tid; j < MT * DD; j += 256) {
    const int i    = j & 15;
    const int ln   = (j >> 4) & 31;
    const int ks   = (j >> 9) & 15;
    const int msub = j >> 13;
    const int row  = msub * 16 + (ln & 15);
    const int k    = ks * 32 + (ln >> 4) * 16 + i;
    const float f  = target[(size_t)(m0 + row) * DD + k];
    const __bf16 h = f32_to_bf16(f);
    Ahi[j] = h;
    Alo[j] = f32_to_bf16(f - (float)h);
  }
  __syncthreads();

  // Running per-lane argmax. Lane (wave, mr) owns columns
  // n0 + wave*32 + s*16 + mr, rows msub*16 + half*8 + r (C-matrix layout).
  float bestv[4][8];
  int   besti[4][8];
#pragma unroll
  for (int a = 0; a < 4; ++a)
#pragma unroll
    for (int r = 0; r < 8; ++r) { bestv[a][r] = -3.4e38f; besti[a][r] = 0; }

  for (int n0 = 0; n0 < KK; n0 += NT) {
    int col[CS];
    const __bf16 *bh_base[CS], *bl_base[CS];
#pragma unroll
    for (int s = 0; s < CS; ++s) {
      col[s] = n0 + wave * (CS * 16) + s * 16 + mr;
      // B fragment (32x16 bf16): lane -> column col[s], elems = 16 consecutive
      // K starting at kk + half*16 ==> contiguous 32B loads from bf16 planes.
      bh_base[s] = cb_hi + (size_t)col[s] * DD + half * 16;
      bl_base[s] = cb_lo + (size_t)col[s] * DD + half * 16;
    }
    v8f acc[4][CS] = {};
#pragma unroll 2
    for (int ks = 0; ks < 16; ++ks) {
      v16bf bh[CS], bl[CS], ah[4], al[4];
#pragma unroll
      for (int s = 0; s < CS; ++s) {
        bh[s] = *(const v16bf*)(bh_base[s] + ks * 32);
        bl[s] = *(const v16bf*)(bl_base[s] + ks * 32);
      }
#pragma unroll
      for (int m = 0; m < 4; ++m) {
        const int fo = ((((m << 4) + ks) << 5) | lane) << 4;
        ah[m] = *(const v16bf*)(Ahi + fo);
        al[m] = *(const v16bf*)(Alo + fo);
      }
      // t.c ~= hi*hi + hi*lo + lo*hi (f32 accumulate) => f32-accurate dot.
      // Issue term-by-term across the 8 independent accumulator chains so
      // consecutive WMMAs never RAW-chain the same accumulator.
#pragma unroll
      for (int m = 0; m < 4; ++m)
#pragma unroll
        for (int s = 0; s < CS; ++s)
          acc[m][s] = __builtin_amdgcn_wmma_f32_16x16x32_bf16(
              false, ah[m], false, bh[s], (short)0, acc[m][s], false, false);
#pragma unroll
      for (int m = 0; m < 4; ++m)
#pragma unroll
        for (int s = 0; s < CS; ++s)
          acc[m][s] = __builtin_amdgcn_wmma_f32_16x16x32_bf16(
              false, ah[m], false, bl[s], (short)0, acc[m][s], false, false);
#pragma unroll
      for (int m = 0; m < 4; ++m)
#pragma unroll
        for (int s = 0; s < CS; ++s)
          acc[m][s] = __builtin_amdgcn_wmma_f32_16x16x32_bf16(
              false, al[m], false, bh[s], (short)0, acc[m][s], false, false);
    }
    // epilogue: add bias, update running best. Strict > keeps the lowest
    // index (jnp.argmin tie behavior) since columns are visited in
    // increasing order (s ascending, n0 ascending).
#pragma unroll
    for (int s = 0; s < CS; ++s) {
      const float bs = bias[col[s]];
#pragma unroll
      for (int m = 0; m < 4; ++m)
#pragma unroll
        for (int r = 0; r < 8; ++r) {
          const float v = acc[m][s][r] + bs;
          if (v > bestv[m][r]) { bestv[m][r] = v; besti[m][r] = col[s]; }
        }
    }
  }

  // Reduce across the 16 columns held by lanes sharing `half`.
#pragma unroll
  for (int m = 0; m < 4; ++m)
#pragma unroll
    for (int r = 0; r < 8; ++r) {
      float v = bestv[m][r];
      int  ix = besti[m][r];
#pragma unroll
      for (int off = 1; off < 16; off <<= 1) {
        const float ov = __shfl_xor(v, off, 32);
        const int   oi = __shfl_xor(ix, off, 32);
        if (ov > v || (ov == v && oi < ix)) { v = ov; ix = oi; }
      }
      if (mr == 0) {
        const int row = m * 16 + half * 8 + r;
        redv[row * 8 + wave] = v;
        redi[row * 8 + wave] = ix;
      }
    }
  __syncthreads();
  if (tid < MT) {
    float v = redv[tid * 8];
    int  ix = redi[tid * 8];
#pragma unroll
    for (int w = 1; w < 8; ++w) {
      const float ov = redv[tid * 8 + w];
      const int   oi = redi[tid * 8 + w];
      if (ov > v || (ov == v && oi < ix)) { v = ov; ix = oi; }
    }
    labels[m0 + tid] = ix;
  }
}

// ---------------------------------------------------------------------------
// Kernel 3: write the full one-hot matrix in a single pass (1 GB stream).
// ---------------------------------------------------------------------------
__global__ void vq_onehot_kernel(const int* __restrict__ labels,
                                 float4* __restrict__ onehot4) {
  const size_t total  = (size_t)NN * (KK / 4);
  const size_t stride = gridDim.x * (size_t)blockDim.x;
  for (size_t j = blockIdx.x * (size_t)blockDim.x + threadIdx.x; j < total;
       j += stride) {
    const int n   = (int)(j >> 11);     // KK/4 = 2048 float4 per row
    const int c4  = (int)(j & 2047);
    const int lab = labels[n];
    float4 v = make_float4(0.f, 0.f, 0.f, 0.f);
    if ((lab >> 2) == c4) ((float*)&v)[lab & 3] = 1.0f;
    onehot4[j] = v;
  }
}

// ---------------------------------------------------------------------------
// Kernel 4: zero codebook-accumulator region of d_out + count scratch
// ---------------------------------------------------------------------------
__global__ void vq_zero_kernel(float4* __restrict__ cb4, size_t n4,
                               float4* __restrict__ cnt4) {
  const size_t i      = blockIdx.x * (size_t)blockDim.x + threadIdx.x;
  const size_t stride = gridDim.x * (size_t)blockDim.x;
  const float4 z = make_float4(0.f, 0.f, 0.f, 0.f);
  for (size_t j = i; j < n4; j += stride) cb4[j] = z;
  if (i < KK / 4) cnt4[i] = z;
}

// ---------------------------------------------------------------------------
// Kernel 5: histogram + segment-sum of targets (atomics). One wave per row.
// ---------------------------------------------------------------------------
__global__ void vq_scatter_kernel(const float* __restrict__ target,
                                  const int* __restrict__ labels,
                                  float* __restrict__ memacc,
                                  float* __restrict__ countw) {
  const int n    = blockIdx.x * 8 + (threadIdx.x >> 5);
  const int lane = threadIdx.x & 31;
  const int lab  = labels[n];
  if (lane == 0) atomicAdd(countw + lab, 1.0f);
  const float* t = target + (size_t)n * DD;
  float* m = memacc + (size_t)lab * DD;
  for (int d = lane; d < DD; d += 32) atomicAdd(m + d, t[d]);
}

// ---------------------------------------------------------------------------
// Kernel 6: EMA blend: new_cb = a*cb + (1-a)*mem ; new_counts likewise
// ---------------------------------------------------------------------------
__global__ void vq_finalize_kernel(const float* __restrict__ codebook,
                                   const float* __restrict__ counts,
                                   const float* __restrict__ countw,
                                   float* __restrict__ out_cb,
                                   float* __restrict__ out_counts) {
  const size_t j = blockIdx.x * (size_t)256 + threadIdx.x;  // < KK*DD
  const int k = (int)(j >> 9);
  const float c = countw[k];
  const float alpha = (c != 0.0f) ? 0.99f : 1.0f;
  out_cb[j] = alpha * codebook[j] + (1.0f - alpha) * out_cb[j];
  if ((j & (DD - 1)) == 0)
    out_counts[k] = alpha * counts[k] + (1.0f - alpha) * c;
}

// ---------------------------------------------------------------------------
extern "C" void kernel_launch(void* const* d_in, const int* in_sizes, int n_in,
                              void* d_out, int out_size, void* d_ws,
                              size_t ws_size, hipStream_t stream) {
  const float* target   = (const float*)d_in[0];
  const float* codebook = (const float*)d_in[1];
  const float* counts   = (const float*)d_in[2];

  // Output layout: [onehot N*K | new_codebook K*D | new_counts K]
  float* onehot  = (float*)d_out;
  float* out_cb  = onehot + (size_t)NN * KK;
  float* out_cnt = out_cb + (size_t)KK * DD;

  // Workspace: bf16 hi/lo code planes (16MB) + bias + labels + count scratch
  char* ws = (char*)d_ws;
  __bf16* cb_hi = (__bf16*)ws;
  __bf16* cb_lo = (__bf16*)(ws + (size_t)KK * DD * 2);
  float*  bias  = (float*)(ws + (size_t)KK * DD * 4);
  int*    labels= (int*)(ws + (size_t)KK * DD * 4 + (size_t)KK * 4);
  float*  countw= (float*)(ws + (size_t)KK * DD * 4 + (size_t)KK * 4 +
                           (size_t)NN * 4);

  vq_prep_kernel<<<KK / 8, 256, 0, stream>>>(codebook, counts, cb_hi, cb_lo,
                                             bias);

  const size_t smem = (size_t)2 * 64 * DD * sizeof(__bf16);  // 128 KB
  vq_gemm_argmax_kernel<<<NN / 64, 256, smem, stream>>>(target, cb_hi, cb_lo,
                                                        bias, labels);

  vq_onehot_kernel<<<32768, 256, 0, stream>>>(labels, (float4*)onehot);

  vq_zero_kernel<<<2048, 256, 0, stream>>>((float4*)out_cb,
                                           (size_t)KK * DD / 4,
                                           (float4*)countw);

  vq_scatter_kernel<<<NN / 8, 256, 0, stream>>>(target, labels, out_cb,
                                                countw);

  vq_finalize_kernel<<<(KK * DD) / 256, 256, 0, stream>>>(codebook, counts,
                                                          countw, out_cb,
                                                          out_cnt);
}
